// kernel_generated_0_62904091017311
// MI455X (gfx1250) — matrix operand order repaired and checked by reading, not compiled, not run on hardware
//
#include <hip/hip_runtime.h>

typedef __attribute__((ext_vector_type(16))) _Float16 v16h;
typedef __attribute__((ext_vector_type(4)))  _Float16 v4h;
typedef __attribute__((ext_vector_type(8)))  float    v8f;
typedef __attribute__((ext_vector_type(4)))  float    v4f;
// 16B-aligned 16-half vector: LDS rows are 16B- but not 32B-aligned.
typedef _Float16 v16hu __attribute__((ext_vector_type(16), aligned(16)));

// LDS row stride (in _Float16) per staged n-row: 60 "op" slots * 64 k + 8 pad.
// 1924 dwords == 4 (mod 64) -> B-operand ds_load_b128s are a clean 2-phase pattern.
#define NSTR 3848

// ---------------------------------------------------------------------------
// Weight prep: pack w[k, j, l, i] (f32) into the exact v_wmma_f32_16x16x32_f16
// A-operand register image (A[i][k] = w[k,j,l,i]) as f16.
// wpack layout: [t(6) = l*3+j][kc(2)][i_tile(4)][lane(32)][e(16)]
// A layout (16-bit A 16x32): lane l: m = l%16, half = l/16,
//   K = half*16 + e (contiguous, matches B); K += kc*32.
// ---------------------------------------------------------------------------
__global__ void prep_weights_62904_kernel(const float* __restrict__ w,
                                          _Float16* __restrict__ wpack) {
  int f    = blockIdx.x * 256 + threadIdx.x;  // 0..1535
  int lane = f & 31;
  int it   = (f >> 5) & 3;
  int kc   = (f >> 7) & 1;
  int t    = f >> 8;                          // 0..5
  int l    = t / 3;
  int j    = t - 3 * l;
  int i    = it * 16 + (lane & 15);
  int half = lane >> 4;
  v16h v;
#pragma unroll
  for (int e = 0; e < 16; ++e) {
    int kl = half * 16 + e;
    int k  = kc * 32 + kl;
    v[e]   = (_Float16)w[((k * 3 + j) * 2 + l) * 64 + i];  // w[k][j][l][i]
  }
  *(v16h*)(wpack + (size_t)f * 16) = v;
}

// ---------------------------------------------------------------------------
// Main kernel: grid = (7 n-blocks, 1024 m), 256 threads (8 waves).
// Stage x[m, :, nb*8..+8, :] -> LDS f16 [n][op][k], with 32B k-groups
// XOR-swizzled by (op>>2)&3, then 6-tap shifted GEMM with
// v_wmma_f32_16x16x32_f16 (f32 accumulate). Each WMMA B operand is ONE
// contiguous 32B LDS region -> single v16h load, one address, no shuffles.
// ---------------------------------------------------------------------------
__global__ void __launch_bounds__(256) conv_wmma_62904_kernel(
    const float* __restrict__ x, const _Float16* __restrict__ wpack,
    float* __restrict__ out) {
  __shared__ _Float16 xh[8 * NSTR];  // 61,568 B

  const int tid = threadIdx.x;
  const int nb  = blockIdx.x;  // 0..6
  const int m   = blockIdx.y;  // 0..1023

  // Zero halo columns op in {0,1,58,59}: op0/1 = conv zero-pad (p=-2,-1),
  // op58 = p=56 zero-pad (also used by the o=0 wrap, j=2), op59 unused.
  // (Swizzle permutes within a column, so zeroing whole columns is exact.)
  {
    unsigned int* xw = (unsigned int*)xh;
    for (int idx = tid; idx < 1024; idx += 256) {
      int n   = idx >> 7;
      int r   = idx & 127;
      int opq = r >> 5;
      int op  = (opq < 2) ? opq : (56 + opq);  // 0,1,58,59
      xw[n * (NSTR / 2) + op * 32 + (r & 31)] = 0u;
    }
  }

  // Stage: coalesced non-temporal float4 reads along o (x is streamed once,
  // keep L2 for output write-combining), f32->f16, transposed store with the
  // 32B-group XOR swizzle to spread transpose-store banks.
  {
    const float* xm = x + (size_t)m * (64 * 56 * 56);
#pragma unroll
    for (int itr = 0; itr < 7; ++itr) {
      int c  = itr * 256 + tid;  // exactly 1792 = 7*256 work items
      int o4 = c % 14;
      int k4 = (c / 14) & 15;
      int n  = c / 224;
      int ng = nb * 8 + n;
      const float* px = xm + (size_t)(k4 * 4) * 3136 + ng * 56 + o4 * 4;
      v4f a0 = __builtin_nontemporal_load((const v4f*)(px));
      v4f a1 = __builtin_nontemporal_load((const v4f*)(px + 3136));
      v4f a2 = __builtin_nontemporal_load((const v4f*)(px + 2 * 3136));
      v4f a3 = __builtin_nontemporal_load((const v4f*)(px + 3 * 3136));
      int g16 = k4 >> 2;        // 32B k-group (4 groups of 16 halves)
      int rem = (k4 & 3) << 2;  // position inside the group
#pragma unroll
      for (int oi = 0; oi < 4; ++oi) {
        int op = o4 * 4 + 2 + oi;
        int sw = (op >> 2) & 3;
        v4h v  = {(_Float16)a0[oi], (_Float16)a1[oi],
                  (_Float16)a2[oi], (_Float16)a3[oi]};
        *(v4h*)(xh + n * NSTR + op * 64 + ((g16 ^ sw) << 4) + rem) = v;
      }
    }
  }
  __syncthreads();

  const int lane = tid & 31;
  const int wv   = tid >> 5;  // wave 0..7
  const int it   = wv & 3;    // i-tile (16 i's)
  const int oh   = wv >> 2;   // o-pair half

  // A operands: register-resident weights for this i-tile (12 x v16h).
  v16h A[12];
#pragma unroll
  for (int t = 0; t < 6; ++t)
#pragma unroll
    for (int kc = 0; kc < 2; ++kc)
      A[t * 2 + kc] = *(const v16h*)(
          wpack + (size_t)(((t * 2 + kc) * 4 + it) * 32 + lane) * 16);

  // B / D lane roles: N-column = (o_off, n_l); D: lane%16 = column,
  // row i = it*16 + (lane/16)*8 + r.
  const int c16  = lane & 15;
  const int nl   = c16 & 7;    // n within block
  const int ooff = c16 >> 3;   // o within the o-pair
  const int half = lane >> 4;  // B: K half (0: K0-15, 1: K16-31)
  const int ng   = nb * 8 + nl;
  const int nout = (ng == 55) ? 0 : (ng + 1);  // roll along n
  const int lbase = nl * NSTR;                 // per-lane LDS row base

  for (int o2 = oh * 14; o2 < oh * 14 + 14; ++o2) {
    const int od  = o2 * 2 + ooff;        // this lane's output o
    const int ob0 = od + 1;               // l=0 taps: op = od+1+j
    const int ob1 = (od == 0) ? 56 : od;  // l=1 taps: op = ob1+j (o=0 wraps to 54/55)
    v8f acc = {};
#pragma unroll
    for (int t = 0; t < 6; ++t) {
      const int op   = (t < 3) ? (ob0 + t) : (ob1 + (t - 3));
      const int sw   = (op >> 2) & 3;
      const int base = lbase + op * 64;
#pragma unroll
      for (int kc = 0; kc < 2; ++kc) {
        const int g16 = kc * 2 + half;  // 32B group holding K[kc*32+half*16 ..+16)
        v16hu bu = *(const v16hu*)(xh + base + ((g16 ^ sw) << 4));
        v16h  b  = (v16h)bu;
        acc = __builtin_amdgcn_wmma_f32_16x16x32_f16(
            false, A[t * 2 + kc], false, b, (short)0, acc, false, false);
      }
    }
    // out[m, i, od, nout]; lanes 0-7 / 8-15 cover contiguous n runs.
    float* po =
        out + ((size_t)(m * 64 + it * 16 + half * 8) * 56 + od) * 56 + nout;
#pragma unroll
    for (int r = 0; r < 8; ++r) po[(size_t)r * 3136] = acc[r];
  }
}

extern "C" void kernel_launch(void* const* d_in, const int* in_sizes, int n_in,
                              void* d_out, int out_size, void* d_ws,
                              size_t ws_size, hipStream_t stream) {
  const float* x = (const float*)d_in[0];  // (1024, 3584, 56) fp32
  const float* w = (const float*)d_in[1];  // (64, 3, 2, 64)  fp32
  // d_in[2] = shift (==1), behavior hardcoded for shift=1.
  _Float16* wpack = (_Float16*)d_ws;       // 49,152 B scratch
  float* out = (float*)d_out;              // (1024, 64, 56, 56) fp32

  prep_weights_62904_kernel<<<6, 256, 0, stream>>>(w, wpack);
  conv_wmma_62904_kernel<<<dim3(7, 1024), 256, 0, stream>>>(x, wpack, out);
}
